// S2HabitFormation_69183333203996
// MI455X (gfx1250) — compile-verified, matrix operand order checked by reading, not run on hardware
//
#include <hip/hip_runtime.h>

// ---------------- problem constants ----------------
#define B_TOTAL   262144
#define STATE_DIM 100
#define KPAD      128      // state K padded to 128 for WMMA
#define HID       256
#define ENC       128
#define EXP_H     64
#define ACT       10
#define NEXP      10
#define ROWS_BLK  128      // rows per block (8 waves x 16 rows)
#define NBLK      (B_TOTAL / ROWS_BLK)   // 2048

typedef __attribute__((ext_vector_type(16))) __bf16 bf16x16;
typedef __attribute__((ext_vector_type(8)))  float  f32x8;

union FragU { bf16x16 bf; unsigned int u[8]; };

// round-to-nearest-even float -> bf16 bits
__device__ __forceinline__ unsigned short f2bf(float f) {
  unsigned int u = __float_as_uint(f);
  unsigned int r = u + 0x7FFFu + ((u >> 16) & 1u);
  return (unsigned short)(r >> 16);
}
__device__ __forceinline__ float bf2f(unsigned short b) {
  return __uint_as_float(((unsigned int)b) << 16);
}

// Load a 16x32 bf16 WMMA fragment from LDS, rows stored K-contiguous.
// ISA 16-bit A/B layout: lane&15 = M (or N); lanes>=16 take K+8 / K+24 halves;
// VGPRs 0..3 = K pairs {kb+0..7}, VGPRs 4..7 = K pairs {kb+16..23}.
__device__ __forceinline__ bf16x16 ldsFrag(const unsigned short* base, int row,
                                           int stride, int kc, int lane) {
  FragU f;
  const unsigned short* p = base + (row + (lane & 15)) * stride
                                 + kc * 32 + ((lane >> 4) << 3);
  const unsigned int* q0 = (const unsigned int*)p;
  const unsigned int* q1 = (const unsigned int*)(p + 16);
  f.u[0] = q0[0]; f.u[1] = q0[1]; f.u[2] = q0[2]; f.u[3] = q0[3];
  f.u[4] = q1[0]; f.u[5] = q1[1]; f.u[6] = q1[2]; f.u[7] = q1[3];
  return f.bf;
}

__device__ __forceinline__ f32x8 wmma_bf16(bf16x16 a, bf16x16 b, f32x8 c) {
  return __builtin_amdgcn_wmma_f32_16x16x32_bf16(false, a, false, b,
                                                 (short)0, c, false, false);
}

// =====================================================================
// Kernel 0: one-time weight repack to bf16 in WMMA-friendly layouts.
//   w1t_bf  [n][kk]      256 x 128   (kk = padded state dim)
//   we1_bf  [k][h][kk]   10 x 64 x 128
//   we2_bf  [k][a16][h]  10 x 16 x 64 (a padded 10->16)
// =====================================================================
#define N_W1T  (HID * KPAD)            // 32768
#define N_WE1  (NEXP * EXP_H * ENC)    // 81920
#define N_WE2  (NEXP * 16 * EXP_H)     // 10240

__global__ __launch_bounds__(256)
void k0_prep(const float* __restrict__ W1, const float* __restrict__ We1,
             const float* __restrict__ We2,
             unsigned short* __restrict__ w1t_bf,
             unsigned short* __restrict__ we1_bf,
             unsigned short* __restrict__ we2_bf) {
  const int total = N_W1T + N_WE1 + N_WE2;
  for (int i = blockIdx.x * 256 + threadIdx.x; i < total;
       i += gridDim.x * 256) {
    if (i < N_W1T) {
      int n = i >> 7, kk = i & (KPAD - 1);
      w1t_bf[i] = f2bf(kk < STATE_DIM ? W1[kk * HID + n] : 0.f);
    } else if (i < N_W1T + N_WE1) {
      int j = i - N_W1T;
      int k = j / (EXP_H * ENC), rem = j % (EXP_H * ENC);
      int h = rem >> 7, kk = rem & (ENC - 1);
      we1_bf[j] = f2bf(We1[((long)k * ENC + kk) * EXP_H + h]);
    } else {
      int j = i - N_W1T - N_WE1;
      int k = j / (16 * EXP_H), rem = j % (16 * EXP_H);
      int a = rem >> 6, h = rem & (EXP_H - 1);
      we2_bf[j] = f2bf(a < ACT ? We2[(k * EXP_H + h) * ACT + a] : 0.f);
    }
  }
}

// =====================================================================
// Kernel 1: per-block WMMA GEMM1 (state@W1+b1, ReLU) -> deterministic
// per-block channel sum / sum-of-squares partials (no FP atomics).
// =====================================================================
struct SmemK1 {
  alignas(16) unsigned short st[ROWS_BLK * KPAD];   // state tile bf16
  alignas(16) unsigned short w1t[HID * KPAD];       // W1^T bf16
  alignas(16) float b1[HID];
  alignas(16) float part[8 * HID * 2];              // per-wave {sum, sumsq}
};

__global__ __launch_bounds__(256)
void k1_stats(const float* __restrict__ state,
              const unsigned short* __restrict__ w1t_bf,
              const float* __restrict__ b1,
              float* __restrict__ pSum, float* __restrict__ pSq) {
  __shared__ SmemK1 sm;
  const int tid  = threadIdx.x;
  const int lane = tid & 31;
  const int wave = tid >> 5;
  const long rowBase = (long)blockIdx.x * ROWS_BLK;

  for (int i = tid; i < ROWS_BLK * KPAD; i += 256) {
    int m = i >> 7, k = i & (KPAD - 1);
    sm.st[i] = f2bf(k < STATE_DIM ? state[(rowBase + m) * STATE_DIM + k] : 0.f);
  }
  {
    const uint4* src = (const uint4*)w1t_bf;
    uint4* dst = (uint4*)sm.w1t;
    for (int i = tid; i < N_W1T / 8; i += 256) dst[i] = src[i];
  }
  if (tid < HID) sm.b1[tid] = b1[tid];
  __syncthreads();

  const int m0 = wave * 16;
  for (int nt = 0; nt < HID / 16; ++nt) {
    f32x8 acc = {};
    for (int kc = 0; kc < KPAD / 32; ++kc)
      acc = wmma_bf16(ldsFrag(sm.st, m0, KPAD, kc, lane),
                      ldsFrag(sm.w1t, nt * 16, KPAD, kc, lane), acc);
    const int c = nt * 16 + (lane & 15);
    const float bias = sm.b1[c];
    float s = 0.f, q = 0.f;
#pragma unroll
    for (int r = 0; r < 8; ++r) {
      float v = acc[r] + bias;
      v = v > 0.f ? v : 0.f;
      s += v; q += v * v;
    }
    s += __shfl_down(s, 16);            // fold lanes 16..31 (rows 8..15)
    q += __shfl_down(q, 16);
    if (lane < 16) {
      sm.part[(wave * HID + c) * 2 + 0] = s;
      sm.part[(wave * HID + c) * 2 + 1] = q;
    }
  }
  __syncthreads();
  if (tid < HID) {                       // fixed-order wave reduction
    float S = 0.f, Q = 0.f;
    for (int w = 0; w < 8; ++w) {
      S += sm.part[(w * HID + tid) * 2 + 0];
      Q += sm.part[(w * HID + tid) * 2 + 1];
    }
    pSum[(long)blockIdx.x * HID + tid] = S;
    pSq [(long)blockIdx.x * HID + tid] = Q;
  }
}

// =====================================================================
// Kernel 2: reduce partials -> mu/var; fold BN affine into W2/b2:
//   W2'[n][k] = scale[k]*W2[k][n]   (bf16, B-fragment-friendly layout)
//   b2'[n]    = b2[n] + sum_k shift[k]*W2[k][n]
// =====================================================================
__global__ __launch_bounds__(256)
void k2_finalize(const float* __restrict__ pSum, const float* __restrict__ pSq,
                 const float* __restrict__ gamma, const float* __restrict__ beta,
                 const float* __restrict__ W2, const float* __restrict__ b2,
                 unsigned short* __restrict__ w2p, float* __restrict__ b2p) {
  __shared__ float sScale[HID], sShift[HID];
  const int t = threadIdx.x;
  {
    float S = 0.f, Q = 0.f;
    for (int b = 0; b < NBLK; ++b) {     // fixed-order block reduction
      S += pSum[(long)b * HID + t];
      Q += pSq [(long)b * HID + t];
    }
    const float invB = 1.0f / (float)B_TOTAL;
    float mu  = S * invB;
    float var = Q * invB - mu * mu;
    var = var > 0.f ? var : 0.f;
    float sc = gamma[t] * rsqrtf(var + 1e-5f);
    sScale[t] = sc;
    sShift[t] = beta[t] - mu * sc;
  }
  __syncthreads();
  for (int i = t; i < ENC * HID; i += 256) {   // [n][k] layout
    int n = i >> 8, k = i & (HID - 1);
    w2p[i] = f2bf(sScale[k] * W2[k * ENC + n]);
  }
  if (t < ENC) {
    float s = b2[t];
    for (int k = 0; k < HID; ++k) s += sShift[k] * W2[k * ENC + t];
    b2p[t] = s;
  }
}

// =====================================================================
// Kernel 3: fully fused forward. Per block: 128 rows, 8 waves.
// LDS overlaid across phases (~215 KB of 320 KB WGP budget).
// =====================================================================
struct SmemK3 {
  union {                                    // 64 KB, phase-dependent
    alignas(16) unsigned short w1t[HID * KPAD];    // phase 1: W1^T [n][k]
    alignas(16) unsigned short w2t[ENC * HID];     // phase 2: W2'^T [n][k]
    alignas(16) unsigned short we1k[EXP_H * ENC];  // phase 4: one expert [h][k]
  } big;
  alignas(16) unsigned short h[ROWS_BLK * HID];    // relu(state@W1+b1), bf16
  union {
    alignas(16) unsigned short st[ROWS_BLK * KPAD];  // phase 1: state tile
    alignas(16) unsigned short enc[ROWS_BLK * 136];  // phase 2+: enc, padded
  } x;
  alignas(16) float att[NEXP * ENC];
  alignas(16) float attn2[NEXP];             // ||attractor_k||^2
  alignas(16) float str[ROWS_BLK * NEXP];
  alignas(16) unsigned short eh[ROWS_BLK * 72];    // expert hidden, padded
  alignas(16) unsigned short we2[NEXP * 16 * EXP_H];
  alignas(16) float b1[HID];
  alignas(16) float be1[NEXP * EXP_H];
  alignas(16) float be2[NEXP * 16];
  alignas(16) float b2p[ENC];
};

__global__ __launch_bounds__(256)
void k3_fused(const float* __restrict__ state,
              const unsigned short* __restrict__ w1t_bf,
              const float* __restrict__ b1,
              const unsigned short* __restrict__ w2p,
              const float* __restrict__ b2p,
              const unsigned short* __restrict__ we1_bf,
              const float* __restrict__ be1,
              const unsigned short* __restrict__ we2_bf,
              const float* __restrict__ be2,
              const float* __restrict__ attr, float* __restrict__ out) {
  __shared__ SmemK3 sm;
  const int tid  = threadIdx.x;
  const int lane = tid & 31;
  const int wave = tid >> 5;
  const long rowBase = (long)blockIdx.x * ROWS_BLK;
  const int m0 = wave * 16;

  // ---- stage persistent small data + phase-1 operands
  for (int i = tid; i < NEXP * ENC; i += 256) sm.att[i] = attr[i];
  for (int i = tid; i < NEXP * EXP_H; i += 256) sm.be1[i] = be1[i];
  for (int i = tid; i < NEXP * 16; i += 256) {
    int k = i >> 4, a = i & 15;
    sm.be2[i] = (a < ACT) ? be2[k * ACT + a] : 0.f;
  }
  if (tid < ENC) sm.b2p[tid] = b2p[tid];
  if (tid < HID) sm.b1[tid] = b1[tid];
  {
    const uint4* s2 = (const uint4*)we2_bf;
    uint4* d2 = (uint4*)sm.we2;
    for (int i = tid; i < N_WE2 / 8; i += 256) d2[i] = s2[i];
    const uint4* s1 = (const uint4*)w1t_bf;
    uint4* d1 = (uint4*)sm.big.w1t;
    for (int i = tid; i < N_W1T / 8; i += 256) d1[i] = s1[i];
  }
  for (int i = tid; i < ROWS_BLK * KPAD; i += 256) {
    int m = i >> 7, k = i & (KPAD - 1);
    sm.x.st[i] = f2bf(k < STATE_DIM ? state[(rowBase + m) * STATE_DIM + k] : 0.f);
  }
  __syncthreads();

  // attractor norms once per block (att is now valid)
  if (tid < NEXP) {
    float a2 = 0.f;
    for (int c = 0; c < ENC; ++c) { float v = sm.att[tid * ENC + c]; a2 += v * v; }
    sm.attn2[tid] = a2;
  }

  // ---- phase 1: h = relu(state @ W1 + b1)  (WMMA, K=128 padded)
  for (int nt = 0; nt < HID / 16; ++nt) {
    f32x8 acc = {};
    for (int kc = 0; kc < KPAD / 32; ++kc)
      acc = wmma_bf16(ldsFrag(sm.x.st, m0, KPAD, kc, lane),
                      ldsFrag(sm.big.w1t, nt * 16, KPAD, kc, lane), acc);
    const int c = nt * 16 + (lane & 15);
    const float bias = sm.b1[c];
    const int mr = m0 + ((lane >> 4) << 3);
#pragma unroll
    for (int r = 0; r < 8; ++r) {
      float v = acc[r] + bias;
      sm.h[(mr + r) * HID + c] = f2bf(v > 0.f ? v : 0.f);
    }
  }
  __syncthreads();

  // ---- swap W1 -> W2' (BN folded) in the big region
  {
    const uint4* src = (const uint4*)w2p;
    uint4* dst = (uint4*)sm.big.w2t;
    for (int i = tid; i < (ENC * HID) / 8; i += 256) dst[i] = src[i];
  }
  __syncthreads();

  // ---- phase 2: enc = tanh(h @ W2' + b2')
  for (int nt = 0; nt < ENC / 16; ++nt) {
    f32x8 acc = {};
    for (int kc = 0; kc < HID / 32; ++kc)
      acc = wmma_bf16(ldsFrag(sm.h, m0, HID, kc, lane),
                      ldsFrag(sm.big.w2t, nt * 16, HID, kc, lane), acc);
    const int c = nt * 16 + (lane & 15);
    const float bias = sm.b2p[c];
    const int mr = m0 + ((lane >> 4) << 3);
#pragma unroll
    for (int r = 0; r < 8; ++r)
      sm.x.enc[(mr + r) * 136 + c] = f2bf(tanhf(acc[r] + bias));
  }
  __syncthreads();

  // ---- phase 3: distances to attractors + softmax strengths (one row/thread)
  if (tid < ROWS_BLK) {
    const unsigned short* e = &sm.x.enc[tid * 136];
    float dot[NEXP];
#pragma unroll
    for (int k = 0; k < NEXP; ++k) dot[k] = 0.f;
    float n2 = 0.f;
    for (int c = 0; c < ENC; ++c) {     // enc read once; att reads broadcast
      float v = bf2f(e[c]);
      n2 += v * v;
#pragma unroll
      for (int k = 0; k < NEXP; ++k) dot[k] += v * sm.att[k * ENC + c];
    }
    float lg[NEXP];
    float mx = -1e30f;
#pragma unroll
    for (int k = 0; k < NEXP; ++k) {
      float d2 = n2 + sm.attn2[k] - 2.f * dot[k];
      d2 = d2 > 0.f ? d2 : 0.f;
      float l = -sqrtf(d2) * 2.0f;      // -dist / TEMP, TEMP = 0.5
      lg[k] = l;
      mx = l > mx ? l : mx;
    }
    float se = 0.f;
#pragma unroll
    for (int k = 0; k < NEXP; ++k) { lg[k] = __expf(lg[k] - mx); se += lg[k]; }
    float inv = 1.f / se;
#pragma unroll
    for (int k = 0; k < NEXP; ++k) sm.str[tid * NEXP + k] = lg[k] * inv;
  }
  __syncthreads();

  // ---- phase 4: 10 experts, WMMA pair each, strength-weighted accumulation
  f32x8 oacc = {};
  for (int k = 0; k < NEXP; ++k) {
    {                                   // stage We1_k (bf16, [h][kk]) via b128
      const uint4* src = (const uint4*)(we1_bf + (size_t)k * EXP_H * ENC);
      uint4* dst = (uint4*)sm.big.we1k;
      for (int i = tid; i < (EXP_H * ENC) / 8; i += 256) dst[i] = src[i];
    }
    if (k + 1 < NEXP)                   // warm L2/L1 for the next expert slab
      __builtin_prefetch(we1_bf + (size_t)(k + 1) * EXP_H * ENC + tid * 32, 0, 3);
    __syncthreads();

    for (int ht = 0; ht < EXP_H / 16; ++ht) {        // eh = relu(enc@We1_k+be1)
      f32x8 acc = {};
      for (int kc = 0; kc < ENC / 32; ++kc)
        acc = wmma_bf16(ldsFrag(sm.x.enc, m0, 136, kc, lane),
                        ldsFrag(sm.big.we1k, ht * 16, ENC, kc, lane), acc);
      const int hh = ht * 16 + (lane & 15);
      const float bias = sm.be1[k * EXP_H + hh];
      const int mr = m0 + ((lane >> 4) << 3);
#pragma unroll
      for (int r = 0; r < 8; ++r) {
        float v = acc[r] + bias;
        sm.eh[(mr + r) * 72 + hh] = f2bf(v > 0.f ? v : 0.f);
      }
    }
    __syncthreads();

    f32x8 acc = {};                                  // habit = eh @ We2_k
    for (int kc = 0; kc < EXP_H / 32; ++kc)
      acc = wmma_bf16(ldsFrag(sm.eh, m0, 72, kc, lane),
                      ldsFrag(&sm.we2[k * 16 * EXP_H], 0, EXP_H, kc, lane), acc);
    const int a = lane & 15;
    const float bias = sm.be2[k * 16 + a];
    const int mr = m0 + ((lane >> 4) << 3);
#pragma unroll
    for (int r = 0; r < 8; ++r) {
      float w = sm.str[(mr + r) * NEXP + k];
      oacc[r] += w * (acc[r] + bias);
    }
    __syncthreads();                                 // before next expert load
  }

  // ---- store out [B, 10]
  {
    const int a = lane & 15;
    const int mr = m0 + ((lane >> 4) << 3);
    if (a < ACT) {
#pragma unroll
      for (int r = 0; r < 8; ++r)
        out[(rowBase + mr + r) * ACT + a] = oacc[r];
    }
  }
}

// =====================================================================
extern "C" void kernel_launch(void* const* d_in, const int* in_sizes, int n_in,
                              void* d_out, int out_size, void* d_ws, size_t ws_size,
                              hipStream_t stream) {
  (void)in_sizes; (void)n_in; (void)out_size; (void)ws_size;
  const float* state = (const float*)d_in[0];
  const float* W1    = (const float*)d_in[1];
  const float* b1    = (const float*)d_in[2];
  const float* gamma = (const float*)d_in[3];
  const float* beta  = (const float*)d_in[4];
  const float* W2    = (const float*)d_in[5];
  const float* b2    = (const float*)d_in[6];
  const float* We1   = (const float*)d_in[7];
  const float* be1   = (const float*)d_in[8];
  const float* We2   = (const float*)d_in[9];
  const float* be2   = (const float*)d_in[10];
  const float* attr  = (const float*)d_in[11];
  float* out = (float*)d_out;

  char* ws = (char*)d_ws;
  float* pSum = (float*)ws;                                        // NBLK*HID
  float* pSq  = pSum + (size_t)NBLK * HID;                         // NBLK*HID
  unsigned short* w2p = (unsigned short*)(pSq + (size_t)NBLK * HID); // ENC*HID
  float* b2p = (float*)(w2p + (size_t)ENC * HID);                  // ENC
  unsigned short* w1t_bf = (unsigned short*)(b2p + ENC);           // N_W1T
  unsigned short* we1_bf = w1t_bf + N_W1T;                         // N_WE1
  unsigned short* we2_bf = we1_bf + N_WE1;                         // N_WE2

  k0_prep    <<<128,  256, 0, stream>>>(W1, We1, We2, w1t_bf, we1_bf, we2_bf);
  k1_stats   <<<NBLK, 256, 0, stream>>>(state, w1t_bf, b1, pSum, pSq);
  k2_finalize<<<1,    256, 0, stream>>>(pSum, pSq, gamma, beta, W2, b2, w2p, b2p);
  k3_fused   <<<NBLK, 256, 0, stream>>>(state, w1t_bf, b1, w2p, b2p,
                                        we1_bf, be1, we2_bf, be2, attr, out);
}
